// Stage1Loss_17523466567855
// MI455X (gfx1250) — matrix operand order repaired and checked by reading, not compiled, not run on hardware
//
#include <hip/hip_runtime.h>
#include <hip/hip_bf16.h>

typedef __attribute__((ext_vector_type(16))) _Float16 v16h;
typedef __attribute__((ext_vector_type(8)))  _Float16 v8h;
typedef __attribute__((ext_vector_type(8)))  float    v8f;

#define N_ROWS 8192
#define DDIM   256
#define BLK    128
#define NB     (N_ROWS / BLK)            // 64
#define NBLOCKS (NB * (NB + 1) / 2)      // 2080
#define ST     264                       // LDS row stride in halves (528B, conflict-free)
#define EPS    1e-8f
#define MARGIN 0.3f

union Frag16 {
    v16h v;
    struct { v8h lo; v8h hi; } h;
};

// ---------------- Pass 1: normalize rows, write f16 Xn ----------------
__global__ void normalize_rows(const float* __restrict__ hidden,
                               _Float16* __restrict__ xh) {
    const int wave = threadIdx.x >> 5;
    const int lane = threadIdx.x & 31;
    const int row  = blockIdx.x * 8 + wave;           // 1024 blocks * 8 rows
    const float* src = hidden + (size_t)row * DDIM;

    float x[8];
    float ss = 0.f;
    #pragma unroll
    for (int t = 0; t < 8; ++t) {
        x[t] = src[lane + 32 * t];
        ss += x[t] * x[t];
    }
    #pragma unroll
    for (int off = 16; off > 0; off >>= 1)
        ss += __shfl_xor(ss, off, 32);

    const float inv = 1.0f / fmaxf(sqrtf(ss), EPS);
    _Float16* dst = xh + (size_t)row * DDIM;
    #pragma unroll
    for (int t = 0; t < 8; ++t)
        dst[lane + 32 * t] = (_Float16)(x[t] * inv);
}

// ---------------- Pass 2: 128x128 Gram block + loss via WMMA ----------------
// 8 waves, each owns a 32x64 tile: 2 A frags x 4 B frags -> 8 WMMAs per K-step
// with only 12 ds_load_b128 (vs 18 for a 16x128 strip).
__global__ void __launch_bounds__(256)
gram_loss_block(const _Float16* __restrict__ xh,
                const int* __restrict__ labels,
                float* __restrict__ partials) {
    extern __shared__ __align__(16) char smem[];
    _Float16* As  = (_Float16*)smem;                  // 128 * ST halves
    _Float16* Bs  = As + BLK * ST;                    // 128 * ST halves
    int*      labI = (int*)(Bs + BLK * ST);           // 128
    int*      labJ = labI + BLK;                      // 128
    float*    red  = (float*)(labJ + BLK);            // 8 (per-wave partials)

    // map linear block id -> upper-triangular (bi, bj), bi <= bj
    int rem = blockIdx.x, bi = 0;
    while (rem >= NB - bi) { rem -= NB - bi; ++bi; }
    const int bj = bi + rem;

    const int rowA0 = bi * BLK;
    const int rowB0 = bj * BLK;

    // cooperative stage: 128 rows x 256 halves each, 16B chunks
    for (int t = threadIdx.x; t < BLK * (DDIM / 8); t += 256) {
        const int r = t >> 5;            // row within panel
        const int c = (t & 31) * 8;      // half offset within row
        *(uint4*)&As[r * ST + c] = *(const uint4*)&xh[(size_t)(rowA0 + r) * DDIM + c];
        *(uint4*)&Bs[r * ST + c] = *(const uint4*)&xh[(size_t)(rowB0 + r) * DDIM + c];
    }
    if (threadIdx.x < BLK)       labI[threadIdx.x] = labels[rowA0 + threadIdx.x];
    else if (threadIdx.x < 256)  labJ[threadIdx.x - BLK] = labels[rowB0 + threadIdx.x - BLK];
    __syncthreads();

    const int wave = threadIdx.x >> 5;
    const int lane = threadIdx.x & 31;
    const int sel  = lane >> 4;          // 0: lanes 0-15, 1: lanes 16-31
    const int nloc = lane & 15;

    const int tmBase = (wave >> 1) * 32; // 4 row strips of 32
    const int tnBase = (wave & 1) * 64;  // 2 col strips of 64

    // A fragment source rows (M = lane%16 within each 16-row tile)
    const _Float16* arow0 = As + (tmBase + 0  + nloc) * ST;
    const _Float16* arow1 = As + (tmBase + 16 + nloc) * ST;
    // B fragment source rows (column N = lane%16 within each 16-col tile)
    const _Float16* brow  = Bs + (tnBase + nloc) * ST;

    v8f acc[2][4] = {};

    #pragma unroll 2
    for (int k0 = 0; k0 < DDIM; k0 += 32) {
        Frag16 a[2];
        // 16-bit A 16x32: lane<16 -> K 0-7 & 16-23; lane>=16 -> K 8-15 & 24-31
        a[0].h.lo = *(const v8h*)&arow0[k0 + sel * 8];
        a[0].h.hi = *(const v8h*)&arow0[k0 + 16 + sel * 8];
        a[1].h.lo = *(const v8h*)&arow1[k0 + sel * 8];
        a[1].h.hi = *(const v8h*)&arow1[k0 + 16 + sel * 8];

        Frag16 b[4];
        #pragma unroll
        for (int tn = 0; tn < 4; ++tn) {
            // dense 16-bit B 32x16: lane<16 -> K 0-7 & 16-23; lane>=16 -> K 8-15 & 24-31
            const _Float16* bp = brow + tn * 16 * ST + k0 + sel * 8;
            b[tn].h.lo = *(const v8h*)bp;
            b[tn].h.hi = *(const v8h*)(bp + 16);
        }

        #pragma unroll
        for (int ti = 0; ti < 2; ++ti)
            #pragma unroll
            for (int tn = 0; tn < 4; ++tn)
                acc[ti][tn] = __builtin_amdgcn_wmma_f32_16x16x32_f16(
                    false, a[ti].v, false, b[tn].v, (short)0, acc[ti][tn],
                    false, false);
    }

    // epilogue: C layout VGPR r -> M = r + 8*sel, N = lane%16
    float lsum = 0.f;
    #pragma unroll
    for (int ti = 0; ti < 2; ++ti) {
        #pragma unroll
        for (int tn = 0; tn < 4; ++tn) {
            const int jloc = tnBase + tn * 16 + nloc;
            const int gj   = rowB0 + jloc;
            const int lj   = labJ[jloc];
            #pragma unroll
            for (int r = 0; r < 8; ++r) {
                const int m  = tmBase + ti * 16 + r + 8 * sel;
                const int gi = rowA0 + m;
                const float s = acc[ti][tn][r];
                const float v = (labI[m] == lj) ? (1.0f - s)
                                                : fmaxf(s - MARGIN, 0.0f);
                lsum += (gj > gi) ? v : 0.0f;
            }
        }
    }

    #pragma unroll
    for (int off = 16; off > 0; off >>= 1)
        lsum += __shfl_xor(lsum, off, 32);
    if (lane == 0) red[wave] = lsum;
    __syncthreads();
    if (threadIdx.x == 0) {
        float bs = 0.f;
        #pragma unroll
        for (int w = 0; w < 8; ++w) bs += red[w];
        partials[blockIdx.x] = bs;
    }
}

// ---------------- Pass 3: deterministic f64 reduction ----------------
__global__ void final_reduce(const float* __restrict__ partials,
                             float* __restrict__ out) {
    __shared__ double sred[256];
    double s = 0.0;
    for (int i = threadIdx.x; i < NBLOCKS; i += 256) s += (double)partials[i];
    sred[threadIdx.x] = s;
    __syncthreads();
    for (int off = 128; off > 0; off >>= 1) {
        if (threadIdx.x < off) sred[threadIdx.x] += sred[threadIdx.x + off];
        __syncthreads();
    }
    if (threadIdx.x == 0) {
        const double count = (double)N_ROWS * (double)(N_ROWS - 1) * 0.5;
        out[0] = (float)(sred[0] / count);
    }
}

extern "C" void kernel_launch(void* const* d_in, const int* in_sizes, int n_in,
                              void* d_out, int out_size, void* d_ws, size_t ws_size,
                              hipStream_t stream) {
    const float* hidden = (const float*)d_in[0];
    const int*   labels = (const int*)d_in[1];
    float*       out    = (float*)d_out;

    _Float16* xh       = (_Float16*)d_ws;                                   // 4 MB
    float*    partials = (float*)((char*)d_ws + (size_t)N_ROWS * DDIM * 2); // 2080 f32

    normalize_rows<<<N_ROWS / 8, 256, 0, stream>>>(hidden, xh);

    const size_t smem = (size_t)(2 * BLK * ST) * sizeof(_Float16)  // A + B panels
                      + (size_t)(2 * BLK) * sizeof(int)            // labels
                      + 8 * sizeof(float);                         // wave partials
    gram_loss_block<<<NBLOCKS, 256, smem, stream>>>(xh, labels, partials);

    final_reduce<<<1, 256, 0, stream>>>(partials, out);
}
